// PointNetFPModule_85358180040892
// MI455X (gfx1250) — matrix operand order repaired and checked by reading, not compiled, not run on hardware
//
#include <hip/hip_runtime.h>
#include <hip/hip_bf16.h>

// ---------------------------------------------------------------------------
// Problem constants (match reference setup_inputs exactly)
// ---------------------------------------------------------------------------
#define BATCH   4
#define NPTS    8192     // N query points
#define MCTR    2048     // M centers
#define CC      256      // centers_features channels
#define CP      64       // points_features channels
#define CT      64       // temb channels
#define CIN     320      // CC + CP
#define OCH     256      // output channels of both convs
#define NGROUPS 8
#define CPG     32       // channels per group (256/8)

typedef __attribute__((ext_vector_type(16))) __bf16 v16bf;
typedef __attribute__((ext_vector_type(8)))  float  v8f;

// ---------------------------------------------------------------------------
// CDNA5 async global->LDS copy (16 bytes per lane) + ASYNCcnt wait.
// Probe round revealed the builtin signature: param0 is a generic pointer to
// a GCC-style 16B int vector. Guarded by __has_builtin with a synchronous
// fallback so the file always compiles.
// ---------------------------------------------------------------------------
#if __has_builtin(__builtin_amdgcn_global_load_async_to_lds_b128)
#define HAVE_ASYNC_LDS 1
#endif

typedef int async_b128_t __attribute__((vector_size(16)));
typedef __attribute__((address_space(3))) async_b128_t* lds_b128_ptr;

__device__ __forceinline__ void async_copy16(const __bf16* __restrict__ g,
                                             __bf16* l) {
#ifdef HAVE_ASYNC_LDS
    __builtin_amdgcn_global_load_async_to_lds_b128(
        (async_b128_t*)(g),
        (lds_b128_ptr)(unsigned int)(uintptr_t)(l),
        0, 0);
#else
    *reinterpret_cast<uint4*>(l) = *reinterpret_cast<const uint4*>(g);
#endif
}

#if __has_builtin(__builtin_amdgcn_s_wait_asynccnt)
#define WAIT_ASYNC(n) __builtin_amdgcn_s_wait_asynccnt(n)
#else
#define WAIT_ASYNC(n) asm volatile("s_wait_asynccnt %0" ::"i"(n) : "memory")
#endif

// 32-byte (16 x bf16) fragment load via two 16B-aligned b128 accesses.
__device__ __forceinline__ v16bf load_frag32(const __bf16* p) {
    union { uint4 u[2]; v16bf v; } t;
    t.u[0] = *reinterpret_cast<const uint4*>(p);
    t.u[1] = *reinterpret_cast<const uint4*>(p + 8);
    return t.v;
}

// ---------------------------------------------------------------------------
// K0: fp32 -> bf16 conversion (weights)
// ---------------------------------------------------------------------------
__global__ void f32_to_bf16_kernel(const float* __restrict__ src,
                                   __bf16* __restrict__ dst, int n) {
    int i = blockIdx.x * blockDim.x + threadIdx.x;
    if (i < n) dst[i] = (__bf16)src[i];
}

// ---------------------------------------------------------------------------
// K1: three-NN (all M=2048 centers staged in LDS, 24 KB of 320 KB WGP LDS)
// ---------------------------------------------------------------------------
__global__ __launch_bounds__(256)
void three_nn_kernel(const float* __restrict__ pc,   // [B,3,N]
                     const float* __restrict__ cc,   // [B,3,M]
                     int* __restrict__ idxOut,       // [B,N,3]
                     float* __restrict__ wOut) {     // [B,N,3]
    __shared__ float cx[MCTR], cy[MCTR], cz[MCTR];
    const int b = blockIdx.y;
    const float* ccb = cc + (size_t)b * 3 * MCTR;
    for (int m = threadIdx.x; m < MCTR; m += blockDim.x) {
        cx[m] = ccb[m];
        cy[m] = ccb[MCTR + m];
        cz[m] = ccb[2 * MCTR + m];
    }
    __syncthreads();

    const int n = blockIdx.x * blockDim.x + threadIdx.x;
    if (n >= NPTS) return;
    const float* pcb = pc + (size_t)b * 3 * NPTS;
    const float px = pcb[n], py = pcb[NPTS + n], pz = pcb[2 * NPTS + n];

    float d0 = 3.4e38f, d1 = 3.4e38f, d2 = 3.4e38f;
    int   i0 = 0, i1 = 0, i2 = 0;
    for (int m = 0; m < MCTR; ++m) {
        const float dx = px - cx[m], dy = py - cy[m], dz = pz - cz[m];
        const float d = dx * dx + dy * dy + dz * dz;
        if (d < d0)      { d2 = d1; i2 = i1; d1 = d0; i1 = i0; d0 = d; i0 = m; }
        else if (d < d1) { d2 = d1; i2 = i1; d1 = d;  i1 = m; }
        else if (d < d2) { d2 = d;  i2 = m; }
    }
    const float r0 = 1.0f / (d0 + 1e-8f);
    const float r1 = 1.0f / (d1 + 1e-8f);
    const float r2 = 1.0f / (d2 + 1e-8f);
    const float inv = 1.0f / (r0 + r1 + r2);
    const size_t o = ((size_t)b * NPTS + n) * 3;
    idxOut[o] = i0; idxOut[o + 1] = i1; idxOut[o + 2] = i2;
    wOut[o] = r0 * inv; wOut[o + 1] = r1 * inv; wOut[o + 2] = r2 * inv;
}

// ---------------------------------------------------------------------------
// K2: gather + weighted sum.  Point-major outputs:
//   xa  [B, N, CIN] bf16  (interp centers_features ++ points_features)
//   outTemb [B, CT, N] fp32 (directly into its d_out slot, channel-major)
// ---------------------------------------------------------------------------
__global__ __launch_bounds__(256)
void gather_interp_kernel(const int* __restrict__ idx, const float* __restrict__ w,
                          const float* __restrict__ cf,   // [B,CC,M]
                          const float* __restrict__ temb, // [B,CT,M]
                          const float* __restrict__ pf,   // [B,CP,N]
                          __bf16* __restrict__ xa,        // [B,N,CIN]
                          float* __restrict__ outTemb) {  // [B,CT,N]
    const int b = blockIdx.y;
    const int n = blockIdx.x * blockDim.x + threadIdx.x;
    if (n >= NPTS) return;
    const size_t o = ((size_t)b * NPTS + n) * 3;
    const int  i0 = idx[o], i1 = idx[o + 1], i2 = idx[o + 2];
    const float w0 = w[o], w1 = w[o + 1], w2 = w[o + 2];

    const float* cfb = cf + (size_t)b * CC * MCTR;
    __bf16* xrow = xa + ((size_t)b * NPTS + n) * CIN;
#pragma unroll 4
    for (int c = 0; c < CC; ++c) {
        const float* r = cfb + (size_t)c * MCTR;
        xrow[c] = (__bf16)(w0 * r[i0] + w1 * r[i1] + w2 * r[i2]);
    }
    const float* tb = temb + (size_t)b * CT * MCTR;
    float* ot = outTemb + (size_t)b * CT * NPTS;
#pragma unroll 4
    for (int c = 0; c < CT; ++c) {
        const float* r = tb + (size_t)c * MCTR;
        ot[(size_t)c * NPTS + n] = w0 * r[i0] + w1 * r[i1] + w2 * r[i2];
    }
    const float* pfb = pf + (size_t)b * CP * NPTS;
#pragma unroll 4
    for (int c = 0; c < CP; ++c)
        xrow[CC + c] = (__bf16)pfb[(size_t)c * NPTS + n];
}

// ---------------------------------------------------------------------------
// K3: bf16 WMMA GEMM (point-major):  Y[b][n][o] = sum_k X[b][n][k] * W[o][k] + bias[o]
// Block tile: 128 points x 64 out-ch.  8 waves (4 along M x 2 along N),
// wave tile 32x32 = 2x2 v_wmma_f32_16x16x32_bf16.
//  - W slab (64 x K) staged once into LDS (async), read as B fragments with
//    zero repacking (K-contiguous per lane).
//  - A (point) tiles double-buffered via global_load_async_to_lds_b128 with
//    K-contiguous layout so each lane's A fragment is one contiguous 32B read.
// ---------------------------------------------------------------------------
#define BKK 32
#define LDA 40            // 32 + 8 pad (80B rows, 16B aligned)
#define LDWMAX 328        // (K=320) + 8 pad

__device__ __forceinline__ void stage_a_tile(const __bf16* __restrict__ Xb,
                                             __bf16* As, int m0, int K, int k0,
                                             int tid) {
    // 128 rows x 32 K per tile; each thread copies two 16B chunks with the
    // K-contiguous layout: src chunk c (8 elems) -> dst c*8
    const int row = tid >> 1;        // 0..127
    const int h   = tid & 1;        // chunk pair {2h, 2h+1}
    const __bf16* g = Xb + (size_t)(m0 + row) * K + k0 + h * 16;
    __bf16* l = &As[row * LDA];
    const int c0 = 2 * h;
    async_copy16(g,     l + c0 * 8);
    async_copy16(g + 8, l + (c0 + 1) * 8);
}

__global__ __launch_bounds__(256)
void gemm_xT_wmma_kernel(const __bf16* __restrict__ Wt,   // [O,K] bf16
                         const __bf16* __restrict__ X,    // [B,N,K] bf16
                         const float* __restrict__ bias,  // [O]
                         float* __restrict__ Y,           // [B,N,O] fp32
                         int O, int K) {
    __shared__ __align__(16) __bf16 Wl[64 * LDWMAX];
    __shared__ __align__(16) __bf16 As[2][128 * LDA];

    const int b = blockIdx.z;
    const __bf16* Xb = X + (size_t)b * NPTS * K;
    float*        Yb = Y + (size_t)b * NPTS * O;
    const int m0 = blockIdx.x * 128;   // point base
    const int o0 = blockIdx.y * 64;    // out-channel base
    const int ldw = K + 8;

    const int tid  = threadIdx.x;
    const int lane = tid & 31;
    const int wave = tid >> 5;
    const int wm = (wave & 3) * 32;    // wave M offset
    const int wn = (wave >> 2) * 32;   // wave N offset

    // stage W slab (64 x K) into LDS once (16B chunks; K multiple of 8)
    for (int idx = tid * 8; idx < 64 * K; idx += 256 * 8) {
        const int row = idx / K, col = idx % K;
        async_copy16(Wt + (size_t)(o0 + row) * K + col, &Wl[row * ldw + col]);
    }
    // stage A tile 0
    stage_a_tile(Xb, As[0], m0, K, 0, tid);

    v8f acc[2][2] = {};
    const int T = K / BKK;
    const int mrow = lane & 15;
    const int kb16 = (lane >> 4) << 4;

    for (int kt = 0; kt < T; ++kt) {
        const int cur = kt & 1;
        if (kt + 1 < T) {
            stage_a_tile(Xb, As[cur ^ 1], m0, K, (kt + 1) * BKK, tid);
            WAIT_ASYNC(2);             // tile kt (and W slab) landed; next tile in flight
        } else {
            WAIT_ASYNC(0);
        }
        __syncthreads();

        const int k0 = kt * BKK;
        v16bf afrag[2], bfrag[2];
        afrag[0] = load_frag32(&As[cur][(wm + mrow) * LDA + kb16]);
        afrag[1] = load_frag32(&As[cur][(wm + 16 + mrow) * LDA + kb16]);
        bfrag[0] = load_frag32(&Wl[(wn + mrow) * ldw + k0 + kb16]);
        bfrag[1] = load_frag32(&Wl[(wn + 16 + mrow) * ldw + k0 + kb16]);

#pragma unroll
        for (int i = 0; i < 2; ++i)
#pragma unroll
            for (int j = 0; j < 2; ++j)
                acc[i][j] = __builtin_amdgcn_wmma_f32_16x16x32_bf16(
                    false, afrag[i], false, bfrag[j],
                    (short)0, acc[i][j], false, false);
        __syncthreads();
    }

    // epilogue: D layout — VGPR r: lanes0-15 (M=r,N=lane), lanes16-31 (M=8+r,N=lane-16)
    const int nOff = lane & 15;
    const int mOff = (lane >> 4) * 8;
#pragma unroll
    for (int j = 0; j < 2; ++j) {
        const int oc = o0 + wn + j * 16 + nOff;
        const float bv = bias[oc];
#pragma unroll
        for (int i = 0; i < 2; ++i) {
#pragma unroll
            for (int r = 0; r < 8; ++r) {
                const int p = m0 + wm + i * 16 + mOff + r;
                Yb[(size_t)p * O + oc] = acc[i][j][r] + bv;   // coalesced along oc
            }
        }
    }
}

// ---------------------------------------------------------------------------
// K4: GroupNorm statistics over point-major y [B,N,256]; one block per (b,g)
// ---------------------------------------------------------------------------
__global__ __launch_bounds__(256)
void gn_stats_kernel(const float* __restrict__ y, float* __restrict__ stats) {
    const int bg = blockIdx.x;
    const int b = bg >> 3, g = bg & 7;
    const float* base = y + (size_t)b * NPTS * OCH + g * CPG;
    const int total = NPTS * CPG;
    float s = 0.f, ss = 0.f;
    for (int i = threadIdx.x; i < total; i += blockDim.x) {
        const float v = base[(size_t)(i >> 5) * OCH + (i & 31)];  // 128B coalesced
        s += v; ss += v * v;
    }
    __shared__ float sh[512];
    sh[threadIdx.x] = s;
    sh[256 + threadIdx.x] = ss;
    __syncthreads();
    for (int o = 128; o > 0; o >>= 1) {
        if (threadIdx.x < o) {
            sh[threadIdx.x]       += sh[threadIdx.x + o];
            sh[256 + threadIdx.x] += sh[256 + threadIdx.x + o];
        }
        __syncthreads();
    }
    if (threadIdx.x == 0) {
        const float mean = sh[0] / (float)total;
        float var = sh[256] / (float)total - mean * mean;
        if (var < 0.f) var = 0.f;
        stats[2 * bg]     = mean;
        stats[2 * bg + 1] = rsqrtf(var + 1e-5f);
    }
}

// ---------------------------------------------------------------------------
// K5a: GN + swish, point-major in/out (layer 1): y [B,N,256] -> bf16 [B,N,256]
// ---------------------------------------------------------------------------
__global__ __launch_bounds__(256)
void gn_swish_nc_kernel(const float* __restrict__ y, const float* __restrict__ stats,
                        const float* __restrict__ gamma, const float* __restrict__ beta,
                        __bf16* __restrict__ outB) {
    const size_t i = (size_t)blockIdx.x * blockDim.x + threadIdx.x;
    const int c = (int)(i & (OCH - 1));
    const int b = (int)(i / ((size_t)NPTS * OCH));
    const int bg = b * NGROUPS + (c >> 5);
    const float mu = stats[2 * bg];
    const float rs = stats[2 * bg + 1];
    const float v = (y[i] - mu) * rs * gamma[c] + beta[c];
    outB[i] = (__bf16)(v / (1.0f + __expf(-v)));
}

// ---------------------------------------------------------------------------
// K5b: GN + swish + transpose (final layer): y [B,N,256] -> out [B,256,N] fp32
// 32-channel x 64-point LDS tile; coalesced on both sides. One block's 32
// channels == exactly one group.
// ---------------------------------------------------------------------------
__global__ __launch_bounds__(256)
void gn_swish_transpose_kernel(const float* __restrict__ y,
                               const float* __restrict__ stats,
                               const float* __restrict__ gamma,
                               const float* __restrict__ beta,
                               float* __restrict__ out) {
    __shared__ float T[32][65];
    const int b  = blockIdx.z;
    const int c0 = blockIdx.y * 32;
    const int n0 = blockIdx.x * 64;
    const int bg = b * NGROUPS + blockIdx.y;
    const float mu = stats[2 * bg];
    const float rs = stats[2 * bg + 1];
    const int t = threadIdx.x;

    {
        const int cr = t & 31;
        const int nr = t >> 5;            // 0..7
        const float ga = gamma[c0 + cr], be = beta[c0 + cr];
#pragma unroll
        for (int it = 0; it < 8; ++it) {
            const int n = nr + it * 8;    // 0..63
            const float v = (y[((size_t)b * NPTS + n0 + n) * OCH + c0 + cr] - mu) * rs * ga + be;
            T[cr][n] = v / (1.0f + __expf(-v));
        }
    }
    __syncthreads();
    {
        const int nw = t & 63;
        const int cw = t >> 6;            // 0..3
#pragma unroll
        for (int it = 0; it < 8; ++it) {
            const int c = cw + it * 4;    // 0..31
            out[((size_t)b * OCH + c0 + c) * NPTS + n0 + nw] = T[c][nw];
        }
    }
}

// ---------------------------------------------------------------------------
// Host launcher
// ---------------------------------------------------------------------------
extern "C" void kernel_launch(void* const* d_in, const int* in_sizes, int n_in,
                              void* d_out, int out_size, void* d_ws, size_t ws_size,
                              hipStream_t stream) {
    (void)in_sizes; (void)n_in; (void)out_size; (void)ws_size;

    const float* pc   = (const float*)d_in[0];
    const float* cc   = (const float*)d_in[1];
    const float* cf   = (const float*)d_in[2];
    const float* pf   = (const float*)d_in[3];
    const float* temb = (const float*)d_in[4];
    const float* W1   = (const float*)d_in[5];
    const float* b1   = (const float*)d_in[6];
    const float* g1   = (const float*)d_in[7];
    const float* be1  = (const float*)d_in[8];
    const float* W2   = (const float*)d_in[9];
    const float* b2   = (const float*)d_in[10];
    const float* g2   = (const float*)d_in[11];
    const float* be2  = (const float*)d_in[12];

    float* out     = (float*)d_out;
    float* outX    = out;                                  // [B,256,N]
    float* outPC   = out + (size_t)BATCH * OCH * NPTS;     // [B,3,N]
    float* outTemb = outPC + (size_t)BATCH * 3 * NPTS;     // [B,CT,N]

    // workspace layout (bytes, 256-aligned)
    char* ws = (char*)d_ws;
    int*    idxW  = (int*)   (ws + 0);                 //   393,216 B
    float*  wW    = (float*) (ws + 393216);            //   393,216 B
    __bf16* xa    = (__bf16*)(ws + 786432);            // 20,971,520 B [B,N,320]
    float*  ybuf  = (float*) (ws + 21757952);          // 33,554,432 B [B,N,256]
    __bf16* x1    = (__bf16*)(ws + 55312384);          // 16,777,216 B [B,N,256]
    float*  stats = (float*) (ws + 72089600);          //        256 B
    __bf16* W1b   = (__bf16*)(ws + 72089856);          //    163,840 B
    __bf16* W2b   = (__bf16*)(ws + 72253696);          //    131,072 B

    // 0) weight conversion
    f32_to_bf16_kernel<<<(OCH * CIN + 255) / 256, 256, 0, stream>>>(W1, W1b, OCH * CIN);
    f32_to_bf16_kernel<<<(OCH * OCH + 255) / 256, 256, 0, stream>>>(W2, W2b, OCH * OCH);

    // pass-through coords output
    (void)hipMemcpyAsync(outPC, pc, (size_t)BATCH * 3 * NPTS * sizeof(float),
                         hipMemcpyDeviceToDevice, stream);

    // 1) three-NN
    three_nn_kernel<<<dim3(NPTS / 256, BATCH), 256, 0, stream>>>(pc, cc, idxW, wW);

    // 2) gather/interpolate (+ temb output, + concat points_features)
    gather_interp_kernel<<<dim3(NPTS / 256, BATCH), 256, 0, stream>>>(
        idxW, wW, cf, temb, pf, xa, outTemb);

    // 3) conv1 GEMM: Y[B,N,256] = xa[B,N,320] x W1^T
    gemm_xT_wmma_kernel<<<dim3(NPTS / 128, OCH / 64, BATCH), 256, 0, stream>>>(
        W1b, xa, b1, ybuf, OCH, CIN);

    // 4) GN1 + swish -> bf16 [B,N,256]
    gn_stats_kernel<<<BATCH * NGROUPS, 256, 0, stream>>>(ybuf, stats);
    gn_swish_nc_kernel<<<(BATCH * OCH * NPTS) / 256, 256, 0, stream>>>(
        ybuf, stats, g1, be1, x1);

    // 5) conv2 GEMM: Y[B,N,256] = x1[B,N,256] x W2^T
    gemm_xT_wmma_kernel<<<dim3(NPTS / 128, OCH / 64, BATCH), 256, 0, stream>>>(
        W2b, x1, b2, ybuf, OCH, OCH);

    // 6) GN2 + swish + transpose -> fp32 [B,256,N] final output
    gn_stats_kernel<<<BATCH * NGROUPS, 256, 0, stream>>>(ybuf, stats);
    gn_swish_transpose_kernel<<<dim3(NPTS / 64, NGROUPS, BATCH), 256, 0, stream>>>(
        ybuf, stats, g2, be2, outX);
}
